// LinearProjectionLayer_17575006175259
// MI455X (gfx1250) — matrix operand order repaired and checked by reading, not compiled, not run on hardware
//
#include <hip/hip_runtime.h>
#include <hip/hip_bf16.h>

// Problem constants (match the reference).
#define DIMK   512         // inner (K) dimension
#define NROWS  65536       // number of rows in x
#define NCOLS  1536        // q(512) + k(512) + v(512) output features

#define LDS_STRIDE 40      // f16 elements per staged row (32 data + 8 pad -> 80B,
                           // row starts sweep all 64 LDS banks, conflict-free)
#define PS (128 * LDS_STRIDE)   // elements per plane per buffer

typedef __attribute__((ext_vector_type(16))) _Float16 v16h;
typedef __attribute__((ext_vector_type(8)))  _Float16 v8h;
typedef __attribute__((ext_vector_type(4)))  _Float16 v4h;
typedef __attribute__((ext_vector_type(8)))  float    v8f;

union Frag16 { v16h v; v8h h[2]; };

// ---------------------------------------------------------------------------
// CDNA5 async copy: global -> LDS, 16B per lane, tracked by ASYNCcnt.
// ---------------------------------------------------------------------------
__device__ __forceinline__ void async_b128(const void* gp, const void* lp) {
    const unsigned lds = (unsigned)(unsigned long long)lp;   // LDS byte offset
    const unsigned long long ga = (unsigned long long)gp;
    asm volatile("global_load_async_to_lds_b128 %0, %1, off"
                 :: "v"(lds), "v"(ga) : "memory");
}
__device__ __forceinline__ void wait_async0() {
    asm volatile("s_wait_asynccnt 0x0" ::: "memory");
}

// ---------------------------------------------------------------------------
// Pass 1: split fp32 -> (hi, lo) _Float16 planes.  f == (float)hi + (float)lo
// to ~22 mantissa bits.
// ---------------------------------------------------------------------------
__global__ __launch_bounds__(256)
void split_planes_kernel(const float* __restrict__ in,
                         _Float16* __restrict__ hi,
                         _Float16* __restrict__ lo,
                         int n) {
    const int i = (blockIdx.x * blockDim.x + threadIdx.x) * 4;
    if (i >= n) return;
    const float4 f = *reinterpret_cast<const float4*>(in + i);
    const float v[4] = {f.x, f.y, f.z, f.w};
    v4h h4, l4;
#pragma unroll
    for (int e = 0; e < 4; ++e) {
        const _Float16 h = (_Float16)v[e];
        h4[e] = h;
        l4[e] = (_Float16)(v[e] - (float)h);
    }
    *reinterpret_cast<v4h*>(hi + i) = h4;
    *reinterpret_cast<v4h*>(lo + i) = l4;
}

// ---------------------------------------------------------------------------
// Fast path: split-f16 WMMA GEMM with double-buffered async-to-LDS staging.
// Block = 8 waves, 128x128 tile, wave = 32x64 (2x4 WMMA tiles).
// acc += xh*wh + xh*wl + xl*wh   (dropped xl*wl ~2^-22 relative)
// ---------------------------------------------------------------------------
__global__ __launch_bounds__(256)
void qkv_wmma_async_kernel(const _Float16* __restrict__ xh,
                           const _Float16* __restrict__ xl,
                           const _Float16* __restrict__ wh,
                           const _Float16* __restrict__ wl,
                           const float* __restrict__ b_q,
                           const float* __restrict__ b_kv,
                           float* __restrict__ out) {
    // Buffers: [2 stages][4 planes: xh, xl, wh, wl][128 rows x LDS_STRIDE]
    __shared__ _Float16 sm[2][4 * PS];

    const int tid  = threadIdx.x;
    const int lane = tid & 31;
    const int wave = tid >> 5;
    const int ln   = lane & 15;     // row (A) / column (B,C,D) index in tile
    const int g    = lane >> 4;     // lane-half selector
    const int rw   = wave & 3;      // row-wave 0..3
    const int cw   = wave >> 2;     // col-wave 0..1

    const int bx = blockIdx.x;      // 0..11 : 128-column block (fastest dim)
    const int by = blockIdx.y;      // 0..511: 128-row block

    // Staging assignment: 2 threads per staged row, 32B (16 f16) each.
    const int rr = tid >> 1;               // 0..127 staged row/col
    const int hh = (tid & 1) * 16;         // element offset within K-chunk
    const size_t xg0 = (size_t)(by * 128 + rr) * DIMK + hh;   // + kc later
    const size_t wg0 = (size_t)(bx * 128 + rr) * DIMK + hh;
    const int    lo0 = rr * LDS_STRIDE + hh;

    // Stage chunk 0 into buffer 0.
    {
        _Float16* sb = &sm[0][0];
        async_b128(xh + xg0,     sb + lo0);
        async_b128(xh + xg0 + 8, sb + lo0 + 8);
        async_b128(xl + xg0,     sb + PS + lo0);
        async_b128(xl + xg0 + 8, sb + PS + lo0 + 8);
        async_b128(wh + wg0,     sb + 2 * PS + lo0);
        async_b128(wh + wg0 + 8, sb + 2 * PS + lo0 + 8);
        async_b128(wl + wg0,     sb + 3 * PS + lo0);
        async_b128(wl + wg0 + 8, sb + 3 * PS + lo0 + 8);
    }

    v8f acc[2][4] = {};

    const int aoff0 = (rw * 32 + ln) * LDS_STRIDE + 8 * g;   // A frag base (r=0)
    const int boff0 = (cw * 64 + ln) * LDS_STRIDE + 8 * g;   // B frag base (c=0)

#pragma unroll 1
    for (int i = 0; i < DIMK / 32; ++i) {
        wait_async0();          // my stage(i) complete
        __syncthreads();        // everyone's stage(i) visible; prev compute done

        if (i + 1 < DIMK / 32) {            // kick off stage(i+1) -> other buf
            const int kc = (i + 1) * 32;
            _Float16* sb = &sm[(i + 1) & 1][0];
            async_b128(xh + xg0 + kc,     sb + lo0);
            async_b128(xh + xg0 + kc + 8, sb + lo0 + 8);
            async_b128(xl + xg0 + kc,     sb + PS + lo0);
            async_b128(xl + xg0 + kc + 8, sb + PS + lo0 + 8);
            async_b128(wh + wg0 + kc,     sb + 2 * PS + lo0);
            async_b128(wh + wg0 + kc + 8, sb + 2 * PS + lo0 + 8);
            async_b128(wl + wg0 + kc,     sb + 3 * PS + lo0);
            async_b128(wl + wg0 + kc + 8, sb + 3 * PS + lo0 + 8);
        }

        const _Float16* sb = &sm[i & 1][0];

        // A fragments (16x32): lane ln = row, half g -> K runs [8g,+8),[16+8g,+8)
        v16h ah[2], al[2];
#pragma unroll
        for (int r = 0; r < 2; ++r) {
            const int ao = aoff0 + 16 * r * LDS_STRIDE;
            Frag16 fh, fl;
            fh.h[0] = *reinterpret_cast<const v8h*>(sb + ao);
            fh.h[1] = *reinterpret_cast<const v8h*>(sb + ao + 16);
            fl.h[0] = *reinterpret_cast<const v8h*>(sb + PS + ao);
            fl.h[1] = *reinterpret_cast<const v8h*>(sb + PS + ao + 16);
            ah[r] = fh.v;
            al[r] = fl.v;
        }

        // B fragments (32x16): lane ln = column, half g -> K runs [8g,+8),[16+8g,+8)
#pragma unroll
        for (int c = 0; c < 4; ++c) {
            const int bo = boff0 + 16 * c * LDS_STRIDE;
            Frag16 fh, fl;
            fh.h[0] = *reinterpret_cast<const v8h*>(sb + 2 * PS + bo);
            fh.h[1] = *reinterpret_cast<const v8h*>(sb + 2 * PS + bo + 16);
            fl.h[0] = *reinterpret_cast<const v8h*>(sb + 3 * PS + bo);
            fl.h[1] = *reinterpret_cast<const v8h*>(sb + 3 * PS + bo + 16);
            const v16h bh = fh.v;
            const v16h bl = fl.v;
#pragma unroll
            for (int r = 0; r < 2; ++r) {
                acc[r][c] = __builtin_amdgcn_wmma_f32_16x16x32_f16(
                    false, ah[r], false, bh, (short)0, acc[r][c], false, false);
                acc[r][c] = __builtin_amdgcn_wmma_f32_16x16x32_f16(
                    false, ah[r], false, bl, (short)0, acc[r][c], false, false);
                acc[r][c] = __builtin_amdgcn_wmma_f32_16x16x32_f16(
                    false, al[r], false, bh, (short)0, acc[r][c], false, false);
            }
        }
    }

    // Bias add + permuted non-temporal store into q/k/v [8, N, 64] planes.
    const int m0 = by * 128 + rw * 32;
    const int j0 = bx * 128 + cw * 64;
#pragma unroll
    for (int c = 0; c < 4; ++c) {
        const int jg = j0 + 16 * c + ln;
        const float bv = (jg < 512) ? b_q[jg] : b_kv[jg - 512];

        int plane, head, d;
        if (jg < 512) {
            plane = 0;            head = jg >> 6;       d = jg & 63;
        } else {
            const int t = jg - 512;
            plane = 1 + (t >> 9); head = (t >> 6) & 7;  d = t & 63;
        }
        const size_t base = (size_t)plane * ((size_t)NROWS * 512)
                          + (size_t)head  * ((size_t)NROWS * 64)
                          + (size_t)d;
#pragma unroll
        for (int r = 0; r < 2; ++r) {
            const int mb = m0 + 16 * r + g * 8;
#pragma unroll
            for (int vi = 0; vi < 8; ++vi) {
                __builtin_nontemporal_store(acc[r][c][vi] + bv,
                                            out + base + (size_t)(mb + vi) * 64);
            }
        }
    }
}

// ---------------------------------------------------------------------------
// Fallback (workspace too small): in-loop conversion version.
// ---------------------------------------------------------------------------
__device__ __forceinline__ void load_split16(const float* __restrict__ p0,
                                             const float* __restrict__ p1,
                                             v16h& hi, v16h& lo) {
    const float4 a0 = *reinterpret_cast<const float4*>(p0);
    const float4 a1 = *reinterpret_cast<const float4*>(p0 + 4);
    const float4 b0 = *reinterpret_cast<const float4*>(p1);
    const float4 b1 = *reinterpret_cast<const float4*>(p1 + 4);
    const float f[16] = {a0.x, a0.y, a0.z, a0.w, a1.x, a1.y, a1.z, a1.w,
                         b0.x, b0.y, b0.z, b0.w, b1.x, b1.y, b1.z, b1.w};
#pragma unroll
    for (int e = 0; e < 16; ++e) {
        const _Float16 h = (_Float16)f[e];
        hi[e] = h;
        lo[e] = (_Float16)(f[e] - (float)h);
    }
}

__global__ __launch_bounds__(256)
void qkv_wmma_fallback_kernel(const float* __restrict__ x,
                              const float* __restrict__ W_q,
                              const float* __restrict__ b_q,
                              const float* __restrict__ W_kv,
                              const float* __restrict__ b_kv,
                              float* __restrict__ out) {
    const int lane = threadIdx.x & 31;
    const int wave = threadIdx.x >> 5;
    const int ln   = lane & 15;
    const int g    = lane >> 4;
    const int rw   = wave & 3;
    const int cw   = wave >> 2;

    const int bx = blockIdx.x;
    const int by = blockIdx.y;

    const bool  is_q = (bx < 4);
    const float* __restrict__ W    = is_q ? W_q : W_kv;
    const float* __restrict__ bias = is_q ? b_q : b_kv;
    const int jloc0 = bx * 128 - (is_q ? 0 : 512);
    const int m0    = by * 128 + rw * 32;
    const int j0    = jloc0 + cw * 64;

    v8f acc[2][4] = {};

    for (int kc = 0; kc < DIMK; kc += 32) {
        v16h ah[2], al[2];
#pragma unroll
        for (int r = 0; r < 2; ++r) {
            const float* rp = x + (size_t)(m0 + 16 * r + ln) * DIMK + kc + g * 8;
            load_split16(rp, rp + 16, ah[r], al[r]);
        }
#pragma unroll
        for (int c = 0; c < 4; ++c) {
            const float* wp = W + (size_t)(j0 + 16 * c + ln) * DIMK + kc + g * 8;
            v16h bh, bl;
            load_split16(wp, wp + 16, bh, bl);
#pragma unroll
            for (int r = 0; r < 2; ++r) {
                acc[r][c] = __builtin_amdgcn_wmma_f32_16x16x32_f16(
                    false, ah[r], false, bh, (short)0, acc[r][c], false, false);
                acc[r][c] = __builtin_amdgcn_wmma_f32_16x16x32_f16(
                    false, ah[r], false, bl, (short)0, acc[r][c], false, false);
                acc[r][c] = __builtin_amdgcn_wmma_f32_16x16x32_f16(
                    false, al[r], false, bh, (short)0, acc[r][c], false, false);
            }
        }
    }

#pragma unroll
    for (int c = 0; c < 4; ++c) {
        const int jloc = j0 + 16 * c + ln;
        const int jg   = jloc + (is_q ? 0 : 512);
        const float bv = bias[jloc];

        int plane, head, d;
        if (jg < 512) {
            plane = 0;            head = jg >> 6;       d = jg & 63;
        } else {
            const int t = jg - 512;
            plane = 1 + (t >> 9); head = (t >> 6) & 7;  d = t & 63;
        }
        const size_t base = (size_t)plane * ((size_t)NROWS * 512)
                          + (size_t)head  * ((size_t)NROWS * 64)
                          + (size_t)d;
#pragma unroll
        for (int r = 0; r < 2; ++r) {
            const int mb = m0 + 16 * r + g * 8;
#pragma unroll
            for (int vi = 0; vi < 8; ++vi) {
                __builtin_nontemporal_store(acc[r][c][vi] + bv,
                                            out + base + (size_t)(mb + vi) * 64);
            }
        }
    }
}

// ---------------------------------------------------------------------------
extern "C" void kernel_launch(void* const* d_in, const int* in_sizes, int n_in,
                              void* d_out, int out_size, void* d_ws, size_t ws_size,
                              hipStream_t stream) {
    (void)in_sizes; (void)n_in; (void)out_size;
    const float* x    = (const float*)d_in[0];
    const float* W_q  = (const float*)d_in[1];
    const float* b_q  = (const float*)d_in[2];
    const float* W_kv = (const float*)d_in[3];
    const float* b_kv = (const float*)d_in[4];
    float* out = (float*)d_out;

    const size_t n_x = (size_t)NROWS * DIMK;   // 33,554,432 elements
    const size_t n_w = (size_t)NCOLS * DIMK;   //    786,432 elements
    const size_t need = (2 * n_x + 2 * n_w) * sizeof(_Float16);  // ~131 MB

    const dim3 grid(NCOLS / 128, NROWS / 128); // column blocks fastest -> L2 reuse
    const dim3 block(256);                     // 8 waves (wave32), 4x2 wave grid

    if (ws_size >= need) {
        _Float16* xh = (_Float16*)d_ws;
        _Float16* xl = xh + n_x;
        _Float16* wh = xl + n_x;
        _Float16* wl = wh + n_w;

        split_planes_kernel<<<(int)(n_x / 1024), 256, 0, stream>>>(x, xh, xl, (int)n_x);
        split_planes_kernel<<<512 * 512 / 1024, 256, 0, stream>>>(W_q, wh, wl, 512 * 512);
        split_planes_kernel<<<1024 * 512 / 1024, 256, 0, stream>>>(
            W_kv, wh + (size_t)512 * 512, wl + (size_t)512 * 512, 1024 * 512);

        qkv_wmma_async_kernel<<<grid, block, 0, stream>>>(xh, xl, wh, wl, b_q, b_kv, out);
    } else {
        qkv_wmma_fallback_kernel<<<grid, block, 0, stream>>>(x, W_q, b_q, W_kv, b_kv, out);
    }
}
